// SimpleGCN_89060441850557
// MI455X (gfx1250) — matrix operand order repaired and checked by reading, not compiled, not run on hardware
//
#include <hip/hip_runtime.h>
#include <hip/hip_bf16.h>

#define HID 64

typedef __attribute__((ext_vector_type(16))) __bf16 v16bf;
typedef __attribute__((ext_vector_type(8)))  float  v8f;

// fp32 -> bf16: clang fptrunc is round-to-nearest-even; gfx1250 has native
// bf16 cvt ops, so this should lower to v_cvt_*bf16* instead of 3 VALU bit-ops.
static __device__ __forceinline__ __bf16 f2bf(float f) {
    return (__bf16)f;
}

// ---------------- degree / normalization ----------------
__global__ void deg_init_kernel(float* deg, int n) {
    int i = blockIdx.x * blockDim.x + threadIdx.x;
    if (i < n) deg[i] = 1.0f;   // self-loop
}

__global__ void deg_count_kernel(const int* __restrict__ dst, float* deg, int nedges) {
    int e = blockIdx.x * blockDim.x + threadIdx.x;
    if (e < nedges) atomicAdd(&deg[dst[e]], 1.0f);
}

__global__ void deg_to_dis_kernel(float* deg, int n) {
    int i = blockIdx.x * blockDim.x + threadIdx.x;
    if (i < n) deg[i] = rsqrtf(deg[i]);   // deg >= 1 always (self loops)
}

__global__ void zero_kernel(float* p, int n) {
    int i = blockIdx.x * blockDim.x + threadIdx.x;
    if (i < n) p[i] = 0.0f;
}

// ---------------- fused (bias+relu) -> bf16 -> WMMA GEMM -> scaled epilogue ----------------
// Computes h = prologue(in) @ W  where prologue = identity (bias==null) or relu(x+bias).
// Writes  hs[r][c]  = h * dis[r]          (pre-scaled messages for the scatter)
//         agg[r][c] = h * dis[r]*dis[r]   (self-loop contribution seeds the aggregation)
// One wave handles a 16x64 output tile; weights (16 KB) hoisted out of the row loop.
__global__ void gcn_gemm_kernel(const float* in,                 // [N,64] (may alias agg)
                                const float* __restrict__ bias,  // [64] or nullptr
                                const float* __restrict__ W,     // [64,64] row-major (k,n)
                                const float* __restrict__ dis,   // [N]
                                float* __restrict__ hs,          // [N,64]
                                float* agg,                      // [N,64] (may alias in)
                                int nrows) {
    const int lane  = threadIdx.x & 31;
    const int half  = lane >> 4;       // 0 or 1
    const int l15   = lane & 15;
    const int gwave = (blockIdx.x * blockDim.x + threadIdx.x) >> 5;
    const int nwave = (gridDim.x * blockDim.x) >> 5;

    // ---- B fragments: bfrag[ntile][kstep]; lane holds col n = 16*ntile+l15,
    //      element v (0..15) = W[kstep*32 + 16*half + v][n]
    v16bf bfrag[4][2];
#pragma unroll
    for (int nt = 0; nt < 4; ++nt)
#pragma unroll
        for (int ks = 0; ks < 2; ++ks) {
            const int n = nt * 16 + l15;
            const int kbase = ks * 32 + half * 16;
#pragma unroll
            for (int v = 0; v < 16; ++v)
                bfrag[nt][ks][v] = f2bf(W[(kbase + v) * HID + n]);
        }

    const bool has_bias = (bias != nullptr);

    for (int rb = gwave; rb * 16 < nrows; rb += nwave) {
        const int arow_i = rb * 16 + l15;        // A row: M = lane&15 for both halves
        const float* arow = in + (size_t)arow_i * HID;

        // ---- A fragments: elems 0..15 = cols ks*32+16*half+0..15
        v16bf afrag[2];
#pragma unroll
        for (int ks = 0; ks < 2; ++ks) {
            const int c0 = ks * 32 + 16 * half;
#pragma unroll
            for (int j = 0; j < 8; ++j) {
                float f0 = arow[c0 + j];
                float f1 = arow[c0 + 8 + j];
                if (has_bias) {
                    f0 = fmaxf(f0 + bias[c0 + j], 0.0f);
                    f1 = fmaxf(f1 + bias[c0 + 8 + j], 0.0f);
                }
                afrag[ks][j]     = f2bf(f0);
                afrag[ks][8 + j] = f2bf(f1);
            }
        }

        // ---- 8x v_wmma_f32_16x16x32_bf16 : 16x64 tile, K=64 in two steps
        v8f acc[4];
#pragma unroll
        for (int nt = 0; nt < 4; ++nt) {
            v8f c = {};
            c = __builtin_amdgcn_wmma_f32_16x16x32_bf16(false, afrag[0], false,
                                                        bfrag[nt][0], (short)0, c,
                                                        false, false);
            c = __builtin_amdgcn_wmma_f32_16x16x32_bf16(false, afrag[1], false,
                                                        bfrag[nt][1], (short)0, c,
                                                        false, false);
            acc[nt] = c;
        }

        // ---- epilogue: row m = r + 8*half, col n = 16*nt + l15
        float dv[8];
#pragma unroll
        for (int r = 0; r < 8; ++r) dv[r] = dis[rb * 16 + r + 8 * half];
#pragma unroll
        for (int nt = 0; nt < 4; ++nt)
#pragma unroll
            for (int r = 0; r < 8; ++r) {
                const int grow = rb * 16 + r + 8 * half;
                const size_t o = (size_t)grow * HID + nt * 16 + l15;
                const float h = acc[nt][r];
                hs[o]  = h * dv[r];
                agg[o] = h * dv[r] * dv[r];
            }
    }
}

// ---------------- edge scatter: agg[dst] += dis[dst] * hs[src] ----------------
// 16 lanes per edge, float4 per lane; hot arrays are L2-resident (52 MB << 192 MB L2),
// so GLOBAL_ATOMIC_ADD_F32 resolves at L2 bandwidth.
__global__ void scatter_kernel(const int* __restrict__ src, const int* __restrict__ dst,
                               const float* __restrict__ dis, const float* __restrict__ hs,
                               float* agg, int nedges) {
    const long long t = (long long)blockIdx.x * blockDim.x + threadIdx.x;
    const long long e = t >> 4;
    const int part = (int)(t & 15);
    if (e >= nedges) return;
    const int s = src[e];
    const int d = dst[e];
    const float w = dis[d];                       // hs already carries dis[src]
    const float4 hv = ((const float4*)(hs + (size_t)s * HID))[part];
    float* out = agg + (size_t)d * HID + part * 4;
    atomicAdd(out + 0, w * hv.x);
    atomicAdd(out + 1, w * hv.y);
    atomicAdd(out + 2, w * hv.z);
    atomicAdd(out + 3, w * hv.w);
}

// ---------------- global mean pool (sum + count) ----------------
__global__ void pool_kernel(const float* __restrict__ agg, const float* __restrict__ bias,
                            const int* __restrict__ batch, float* sums, float* cnts, int n) {
    const long long i = (long long)blockIdx.x * blockDim.x + threadIdx.x;
    if (i >= (long long)n * HID) return;
    const int node = (int)(i >> 6);
    const int col  = (int)(i & 63);
    const float v = fmaxf(agg[i] + bias[col], 0.0f);
    const int g = batch[node];
    atomicAdd(&sums[(size_t)g * HID + col], v);
    if (col == 0) atomicAdd(&cnts[g], 1.0f);
}

// ---------------- classifier head: one wave per graph ----------------
__global__ void classifier_kernel(const float* __restrict__ sums, const float* __restrict__ cnts,
                                  const float* __restrict__ Wc1, const float* __restrict__ bc1,
                                  const float* __restrict__ Wc2, const float* __restrict__ bc2,
                                  float* out, int ngraphs) {
    const int wv = (blockIdx.x * blockDim.x + threadIdx.x) >> 5;
    const int lane = threadIdx.x & 31;
    if (wv >= ngraphs) return;
    const float inv = 1.0f / fmaxf(cnts[wv], 1.0f);
    const float* p = sums + (size_t)wv * HID;
    float z = bc1[lane];                           // lane = hidden unit j (32 units)
#pragma unroll
    for (int k = 0; k < HID; ++k) z += (p[k] * inv) * Wc1[k * 32 + lane];
    z = fmaxf(z, 0.0f);
    float contrib = z * Wc2[lane];                 // Wc2 is [32,1]
#pragma unroll
    for (int off = 16; off > 0; off >>= 1) contrib += __shfl_xor(contrib, off, 32);
    if (lane == 0) out[wv] = contrib + bc2[0];
}

extern "C" void kernel_launch(void* const* d_in, const int* in_sizes, int n_in,
                              void* d_out, int out_size, void* d_ws, size_t ws_size,
                              hipStream_t stream) {
    const float* x    = (const float*)d_in[0];
    const int*   edge = (const int*)d_in[1];
    const int*   batch= (const int*)d_in[2];
    const float* W1 = (const float*)d_in[3];  const float* b1 = (const float*)d_in[4];
    const float* W2 = (const float*)d_in[5];  const float* b2 = (const float*)d_in[6];
    const float* W3 = (const float*)d_in[7];  const float* b3 = (const float*)d_in[8];
    const float* Wc1= (const float*)d_in[9];  const float* bc1= (const float*)d_in[10];
    const float* Wc2= (const float*)d_in[11]; const float* bc2= (const float*)d_in[12];
    float* out = (float*)d_out;

    const int N = in_sizes[0] / HID;          // 100000
    const int E = in_sizes[1] / 2;            // 1600000
    const int G = out_size;                   // 1024
    const int* e_src = edge;
    const int* e_dst = edge + E;

    // workspace partition (all fp32; ~52 MB total, L2-resident)
    float* dis  = (float*)d_ws;               // [N]   (deg, then rsqrt in place)
    float* hs   = dis + N;                    // [N,64]
    float* agg  = hs + (size_t)N * HID;       // [N,64]
    float* sums = agg + (size_t)N * HID;      // [G,64]
    float* cnts = sums + (size_t)G * HID;     // [G]

    const int B = 256;
    // --- normalization ---
    deg_init_kernel<<<(N + B - 1) / B, B, 0, stream>>>(dis, N);
    zero_kernel<<<(G * HID + G + B - 1) / B, B, 0, stream>>>(sums, G * HID + G);
    deg_count_kernel<<<(E + B - 1) / B, B, 0, stream>>>(e_dst, dis, E);
    deg_to_dis_kernel<<<(N + B - 1) / B, B, 0, stream>>>(dis, N);

    const int gemmGrid = 416;                 // 3328 waves, grid-stride over 6250 row tiles
    const long long st = (long long)E * 16;
    const int scatGrid = (int)((st + B - 1) / B);

    // --- layer 1 ---
    gcn_gemm_kernel<<<gemmGrid, B, 0, stream>>>(x,   nullptr, W1, dis, hs, agg, N);
    scatter_kernel<<<scatGrid, B, 0, stream>>>(e_src, e_dst, dis, hs, agg, E);
    // --- layer 2 (bias1+relu fused into prologue) ---
    gcn_gemm_kernel<<<gemmGrid, B, 0, stream>>>(agg, b1,      W2, dis, hs, agg, N);
    scatter_kernel<<<scatGrid, B, 0, stream>>>(e_src, e_dst, dis, hs, agg, E);
    // --- layer 3 ---
    gcn_gemm_kernel<<<gemmGrid, B, 0, stream>>>(agg, b2,      W3, dis, hs, agg, N);
    scatter_kernel<<<scatGrid, B, 0, stream>>>(e_src, e_dst, dis, hs, agg, E);

    // --- mean pool (bias3+relu fused) ---
    const long long pt = (long long)N * HID;
    pool_kernel<<<(int)((pt + B - 1) / B), B, 0, stream>>>(agg, b3, batch, sums, cnts, N);

    // --- classifier head ---
    classifier_kernel<<<(G * 32 + B - 1) / B, B, 0, stream>>>(sums, cnts, Wc1, bc1, Wc2, bc2, out, G);
}
